// GPT2Attention_73134703116714
// MI455X (gfx1250) — matrix operand order repaired and checked by reading, not compiled, not run on hardware
//
#include <hip/hip_runtime.h>
#include <hip/hip_bf16.h>

typedef __bf16 bf16;
typedef __attribute__((ext_vector_type(16))) __bf16 v16bf;
typedef __attribute__((ext_vector_type(8)))  __bf16 v8bf;
typedef __attribute__((ext_vector_type(8)))  float   v8f;

#define B_  4
#define S_  2048
#define D_  1024
#define H_  16
#define HD_ 64

// ---------- WMMA operand loaders (contiguous k = 16h+s for A and B, not the ISA table order; wave32) ----------
// A-matrix 16x32 bf16: lane L holds row M=L&15.
//   lane<16 : elems 0..15 = K+0..15
//   lane>=16: elems 0..15 = K+16..31
__device__ __forceinline__ v16bf load_a16(const bf16* __restrict__ base, long row, int ld,
                                          int k0, int lane) {
    const bf16* p = base + row * (long)ld + k0 + ((lane < 16) ? 0 : 16);
    v8bf lo = *reinterpret_cast<const v8bf*>(p);
    v8bf hi = *reinterpret_cast<const v8bf*>(p + 8);
    v16bf a;
#pragma unroll
    for (int i = 0; i < 8; ++i) { a[i] = lo[i]; a[i + 8] = hi[i]; }
    return a;
}

// ---------- conversion / transpose ----------
__global__ __launch_bounds__(256) void k_cvt_bf16(const float* __restrict__ in,
                                                  bf16* __restrict__ out, long n) {
    long i = (long)blockIdx.x * 256 + threadIdx.x;
    if (i < n) out[i] = (bf16)in[i];
}

// WT[n*D + k] = (bf16) W[k*D + n]
__global__ __launch_bounds__(256) void k_transpose_w(const float* __restrict__ W,
                                                     bf16* __restrict__ WT) {
    long i = (long)blockIdx.x * 256 + threadIdx.x;
    if (i >= (long)D_ * D_) return;
    int n = (int)(i / D_);
    int k = (int)(i % D_);
    WT[i] = (bf16)W[(long)k * D_ + n];
}

// ---------- fused QKV GEMM, 64x64 output per wave (4x4 WMMA tiles) ----------
// 16 WMMAs per 32-deep k-step from 4 A-loads + 4 B-loads -> 32 flops/byte feed.
__global__ __launch_bounds__(256)
void k_gemm_qkv(const bf16* __restrict__ xb,
                const bf16* __restrict__ WqT, const bf16* __restrict__ WkT,
                const bf16* __restrict__ WvT,
                const float* __restrict__ bq, const float* __restrict__ bk,
                const float* __restrict__ bv,
                bf16* __restrict__ Q, bf16* __restrict__ K, bf16* __restrict__ Vt) {
    const int lane = threadIdx.x & 31;
    const int wave = threadIdx.x >> 5;
    const long tilesPerMat = (long)(B_ * S_ / 64) * (D_ / 64);   // 128*16 = 2048
    long tile = (long)blockIdx.x * 8 + wave;
    if (tile >= 3 * tilesPerMat) return;
    const int  mat  = (int)(tile / tilesPerMat);
    const long t    = tile % tilesPerMat;
    const int  tn64 = (int)(t % (D_ / 64));
    const long tm64 = t / (D_ / 64);

    const bf16*  WT   = (mat == 0) ? WqT : (mat == 1) ? WkT : WvT;
    const float* bias = (mat == 0) ? bq  : (mat == 1) ? bk  : bv;

    const int nlo = lane & 15;

    v8f acc[4][4];
#pragma unroll
    for (int ni = 0; ni < 4; ++ni) {
        const float bval = bias[tn64 * 64 + ni * 16 + nlo];
#pragma unroll
        for (int mi = 0; mi < 4; ++mi)
#pragma unroll
            for (int r = 0; r < 8; ++r) acc[mi][ni][r] = bval;
    }

    // B-operand row pointers: lane holds column n, 16 contiguous K per half-wave.
    const bf16* wrow[4];
#pragma unroll
    for (int ni = 0; ni < 4; ++ni)
        wrow[ni] = WT + (long)(tn64 * 64 + ni * 16 + nlo) * D_ + ((lane < 16) ? 0 : 16);
    const long rowbase = tm64 * 64 + nlo;

#pragma unroll 1
    for (int k0 = 0; k0 < D_; k0 += 32) {
        v16bf a[4], b[4];
#pragma unroll
        for (int mi = 0; mi < 4; ++mi) a[mi] = load_a16(xb, rowbase + mi * 16, D_, k0, lane);
#pragma unroll
        for (int ni = 0; ni < 4; ++ni) {
            b[ni] = *reinterpret_cast<const v16bf*>(wrow[ni] + k0);
            __builtin_prefetch(wrow[ni] + k0 + 128, 0, 3);
        }
#pragma unroll
        for (int mi = 0; mi < 4; ++mi)
#pragma unroll
            for (int ni = 0; ni < 4; ++ni)
                acc[mi][ni] = __builtin_amdgcn_wmma_f32_16x16x32_bf16(
                    false, a[mi], false, b[ni], (short)0, acc[mi][ni], false, false);
    }

    // N-tile of 64 aligns exactly with one head: h = tn64.
    const int h = tn64;
#pragma unroll
    for (int mi = 0; mi < 4; ++mi)
#pragma unroll
        for (int r = 0; r < 8; ++r) {
            int  m   = mi * 16 + r + ((lane < 16) ? 0 : 8);
            long row = tm64 * 64 + m;              // = b*S + q
            int  bb  = (int)(row / S_);
            int  q   = (int)(row % S_);
#pragma unroll
            for (int ni = 0; ni < 4; ++ni) {
                int  d   = ni * 16 + nlo;
                bf16 val = (bf16)acc[mi][ni][r];
                if (mat == 2) {
                    // V stored transposed per head: [B,H,HD,S]
                    Vt[((long)(bb * H_ + h) * HD_ + d) * S_ + q] = val;
                } else {
                    bf16* O = (mat == 0) ? Q : K;
                    O[((long)(bb * H_ + h) * S_ + q) * HD_ + d] = val;
                }
            }
        }
}

// ---------- flash attention: wave per (b,h,q-tile of 16 rows) ----------
__global__ __launch_bounds__(128)
void k_attn(const bf16* __restrict__ Q, const bf16* __restrict__ K,
            const bf16* __restrict__ Vt, bf16* __restrict__ Obuf) {
    __shared__ bf16 pbuf[4][16 * 32];
    const int lane = threadIdx.x & 31;
    const int wave = threadIdx.x >> 5;
    const int wid  = blockIdx.x * 4 + wave;        // 0 .. B*H*S/16-1
    const int qt   = wid & (S_ / 16 - 1);
    const int bh   = wid >> 7;                      // b*H + h
    const int q0   = qt * 16;

    const bf16* Qh = Q  + (long)bh * S_ * HD_;
    const bf16* Kh = K  + (long)bh * S_ * HD_;
    const bf16* Vh = Vt + (long)bh * HD_ * S_;
    bf16* lds = pbuf[wave];

    const int nlo  = lane & 15;
    const int mrow = (lane < 16) ? 0 : 8;

    // Q tile as A-operands for the two 32-wide d-chunks; kept in registers.
    v16bf qa[2];
#pragma unroll
    for (int c = 0; c < 2; ++c) qa[c] = load_a16(Qh, q0 + nlo, HD_, c * 32, lane);

    v8f o[4];
#pragma unroll
    for (int dt = 0; dt < 4; ++dt)
#pragma unroll
        for (int r = 0; r < 8; ++r) o[dt][r] = 0.0f;

    float m_run[8], l_run[8];
#pragma unroll
    for (int r = 0; r < 8; ++r) { m_run[r] = -1e30f; l_run[r] = 0.0f; }

    const float sc = 0.125f;                        // 1/sqrt(HD)

    for (int kb = 0; kb < q0 + 16; kb += 32) {
        // ---- scores: two 16x16 sub-tiles over 32 keys, K-dim = HD = 64 ----
        v8f s[2];
#pragma unroll
        for (int sub = 0; sub < 2; ++sub) {
            v8f c;
#pragma unroll
            for (int r = 0; r < 8; ++r) c[r] = 0.0f;
            const bf16* kb_ptr =
                Kh + (long)(kb + sub * 16 + nlo) * HD_ + ((lane < 16) ? 0 : 16);
#pragma unroll
            for (int ch = 0; ch < 2; ++ch) {
                v16bf b = *reinterpret_cast<const v16bf*>(kb_ptr + ch * 32);
                c = __builtin_amdgcn_wmma_f32_16x16x32_bf16(false, qa[ch], false, b,
                                                            (short)0, c, false, false);
            }
            s[sub] = c;
        }
        // ---- scale + causal mask (bias = -10000 like the reference) ----
#pragma unroll
        for (int sub = 0; sub < 2; ++sub) {
            int kg = kb + sub * 16 + nlo;
#pragma unroll
            for (int r = 0; r < 8; ++r) {
                float v = s[sub][r] * sc;
                s[sub][r] = (kg > q0 + mrow + r) ? -10000.0f : v;
            }
        }
        // ---- online softmax (rows live per half-wave; reduce over 16 lanes) ----
#pragma unroll
        for (int r = 0; r < 8; ++r) {
            float tmax = fmaxf(s[0][r], s[1][r]);
#pragma unroll
            for (int off = 1; off < 16; off <<= 1)
                tmax = fmaxf(tmax, __shfl_xor(tmax, off, 32));
            float nm    = fmaxf(m_run[r], tmax);
            float scale = __expf(m_run[r] - nm);
            m_run[r]    = nm;
            float p0 = __expf(s[0][r] - nm);
            float p1 = __expf(s[1][r] - nm);
            s[0][r] = p0; s[1][r] = p1;
            float ts = p0 + p1;
#pragma unroll
            for (int off = 1; off < 16; off <<= 1) ts += __shfl_xor(ts, off, 32);
            l_run[r] = l_run[r] * scale + ts;
#pragma unroll
            for (int dt = 0; dt < 4; ++dt) o[dt][r] *= scale;
        }
        // ---- re-lay P (16x32) through per-wave LDS into A-operand form ----
#pragma unroll
        for (int sub = 0; sub < 2; ++sub)
#pragma unroll
            for (int r = 0; r < 8; ++r)
                lds[(mrow + r) * 32 + sub * 16 + nlo] = (bf16)s[sub][r];
        __builtin_amdgcn_wave_barrier();   // LDS is in-order per wave; block reordering
        v16bf pa;
        {
            const bf16* p = lds + nlo * 32 + ((lane < 16) ? 0 : 16);
            v8bf lo = *reinterpret_cast<const v8bf*>(p);
            v8bf hi = *reinterpret_cast<const v8bf*>(p + 8);
#pragma unroll
            for (int i = 0; i < 8; ++i) { pa[i] = lo[i]; pa[i + 8] = hi[i]; }
        }
        __builtin_amdgcn_wave_barrier();
        // ---- O += P x V^T (B-operand contiguous thanks to transposed V) ----
#pragma unroll
        for (int dt = 0; dt < 4; ++dt) {
            const bf16* vb =
                Vh + (long)(dt * 16 + nlo) * S_ + kb + ((lane < 16) ? 0 : 16);
            v16bf b = *reinterpret_cast<const v16bf*>(vb);
            o[dt] = __builtin_amdgcn_wmma_f32_16x16x32_bf16(false, pa, false, b,
                                                            (short)0, o[dt], false, false);
        }
    }

    // ---- normalize + merge heads: Obuf[b, q, h*64+d] (bf16) ----
    const int bb = bh >> 4;
    const int h  = bh & 15;
#pragma unroll
    for (int r = 0; r < 8; ++r) {
        float inv = 1.0f / l_run[r];
        int   q   = q0 + mrow + r;
#pragma unroll
        for (int dt = 0; dt < 4; ++dt) {
            int d = dt * 16 + nlo;
            Obuf[((long)bb * S_ + q) * D_ + h * 64 + d] = (bf16)(o[dt][r] * inv);
        }
    }
}

// ---------- output projection + bias + residual -> y (f32), 64x64 per wave ----------
__global__ __launch_bounds__(256)
void k_gemm_proj(const bf16* __restrict__ Ob, const bf16* __restrict__ WpT,
                 const float* __restrict__ bp, const float* __restrict__ x,
                 float* __restrict__ y) {
    const int lane = threadIdx.x & 31;
    const int wave = threadIdx.x >> 5;
    const long total = (long)(B_ * S_ / 64) * (D_ / 64);   // 2048
    long tile = (long)blockIdx.x * 8 + wave;
    if (tile >= total) return;
    const int  tn64 = (int)(tile % (D_ / 64));
    const long tm64 = tile / (D_ / 64);
    const int  nlo  = lane & 15;

    v8f acc[4][4];
#pragma unroll
    for (int ni = 0; ni < 4; ++ni) {
        const float bval = bp[tn64 * 64 + ni * 16 + nlo];
#pragma unroll
        for (int mi = 0; mi < 4; ++mi)
#pragma unroll
            for (int r = 0; r < 8; ++r) acc[mi][ni][r] = bval;
    }

    const bf16* wrow[4];
#pragma unroll
    for (int ni = 0; ni < 4; ++ni)
        wrow[ni] = WpT + (long)(tn64 * 64 + ni * 16 + nlo) * D_ + ((lane < 16) ? 0 : 16);
    const long rowbase = tm64 * 64 + nlo;

#pragma unroll 1
    for (int k0 = 0; k0 < D_; k0 += 32) {
        v16bf a[4], b[4];
#pragma unroll
        for (int mi = 0; mi < 4; ++mi) a[mi] = load_a16(Ob, rowbase + mi * 16, D_, k0, lane);
#pragma unroll
        for (int ni = 0; ni < 4; ++ni) {
            b[ni] = *reinterpret_cast<const v16bf*>(wrow[ni] + k0);
            __builtin_prefetch(wrow[ni] + k0 + 128, 0, 3);
        }
#pragma unroll
        for (int mi = 0; mi < 4; ++mi)
#pragma unroll
            for (int ni = 0; ni < 4; ++ni)
                acc[mi][ni] = __builtin_amdgcn_wmma_f32_16x16x32_bf16(
                    false, a[mi], false, b[ni], (short)0, acc[mi][ni], false, false);
    }

#pragma unroll
    for (int mi = 0; mi < 4; ++mi)
#pragma unroll
        for (int r = 0; r < 8; ++r) {
            long row = tm64 * 64 + mi * 16 + r + ((lane < 16) ? 0 : 8);
#pragma unroll
            for (int ni = 0; ni < 4; ++ni) {
                long idx = row * D_ + tn64 * 64 + ni * 16 + nlo;
                y[idx] = acc[mi][ni][r] + x[idx];   // bias already in acc
            }
        }
}

// ---------- LayerNorm over last dim, block per row ----------
__global__ __launch_bounds__(256)
void k_layernorm(const float* __restrict__ y, const float* __restrict__ g,
                 const float* __restrict__ bta, float* __restrict__ out) {
    __shared__ float s1[256], s2[256];
    const int  tid = threadIdx.x;
    const long row = blockIdx.x;
    const float* yr = y + row * D_;
    float sum = 0.0f, sq = 0.0f;
    for (int i = tid; i < D_; i += 256) { float v = yr[i]; sum += v; sq += v * v; }
    s1[tid] = sum; s2[tid] = sq;
    __syncthreads();
    for (int st = 128; st > 0; st >>= 1) {
        if (tid < st) { s1[tid] += s1[tid + st]; s2[tid] += s2[tid + st]; }
        __syncthreads();
    }
    const float mu   = s1[0] * (1.0f / D_);
    const float var  = s2[0] * (1.0f / D_) - mu * mu;
    const float rinv = rsqrtf(var + 1e-5f);
    for (int i = tid; i < D_; i += 256)
        out[row * D_ + i] = (yr[i] - mu) * rinv * g[i] + bta[i];
}

extern "C" void kernel_launch(void* const* d_in, const int* in_sizes, int n_in,
                              void* d_out, int out_size, void* d_ws, size_t ws_size,
                              hipStream_t stream) {
    (void)in_sizes; (void)n_in; (void)out_size; (void)ws_size;
    const float* x   = (const float*)d_in[0];
    const float* Wq  = (const float*)d_in[1];
    const float* bq  = (const float*)d_in[2];
    const float* Wk  = (const float*)d_in[3];
    const float* bk  = (const float*)d_in[4];
    const float* Wv  = (const float*)d_in[5];
    const float* bv  = (const float*)d_in[6];
    const float* Wp  = (const float*)d_in[7];
    const float* bp  = (const float*)d_in[8];
    const float* lng = (const float*)d_in[9];
    const float* lnb = (const float*)d_in[10];
    float* out = (float*)d_out;

    const long NX = (long)B_ * S_ * D_;           // 8,388,608
    const long NW = (long)D_ * D_;                // 1,048,576
    char* ws = (char*)d_ws;
    size_t off = 0;
    bf16* WqT = (bf16*)(ws + off); off += NW * sizeof(bf16);
    bf16* WkT = (bf16*)(ws + off); off += NW * sizeof(bf16);
    bf16* WvT = (bf16*)(ws + off); off += NW * sizeof(bf16);
    bf16* WpT = (bf16*)(ws + off); off += NW * sizeof(bf16);
    bf16* Qb  = (bf16*)(ws + off); off += NX * sizeof(bf16);
    bf16* Kb  = (bf16*)(ws + off); off += NX * sizeof(bf16);
    bf16* Vtb = (bf16*)(ws + off); off += NX * sizeof(bf16);
    bf16* Ob  = (bf16*)(ws + off); off += NX * sizeof(bf16);
    // y (f32, 32MB) aliases the x-bf16 region: xb is dead once QKV GEMM finishes.
    float* y  = (float*)(ws + off);
    bf16*  xb = (bf16*)(ws + off);

    // 1) precision conversion + weight transposes
    k_cvt_bf16<<<(int)((NX + 255) / 256), 256, 0, stream>>>(x, xb, NX);
    k_transpose_w<<<(int)((NW + 255) / 256), 256, 0, stream>>>(Wq, WqT);
    k_transpose_w<<<(int)((NW + 255) / 256), 256, 0, stream>>>(Wk, WkT);
    k_transpose_w<<<(int)((NW + 255) / 256), 256, 0, stream>>>(Wv, WvT);
    k_transpose_w<<<(int)((NW + 255) / 256), 256, 0, stream>>>(Wp, WpT);

    // 2) QKV projections: 3 * (B*S/64) * (D/64) wave-tiles (64x64 each), 8 waves/block
    {
        long tiles = 3L * (B_ * S_ / 64) * (D_ / 64);
        k_gemm_qkv<<<(int)((tiles + 7) / 8), 256, 0, stream>>>(
            xb, WqT, WkT, WvT, bq, bk, bv, Qb, Kb, Vtb);
    }
    // 3) causal flash attention: B*H*(S/16) waves, 4 waves / block
    k_attn<<<(B_ * H_ * (S_ / 16)) / 4, 128, 0, stream>>>(Qb, Kb, Vtb, Ob);

    // 4) output projection + residual (64x64 per wave)
    {
        long tiles = (long)(B_ * S_ / 64) * (D_ / 64);
        k_gemm_proj<<<(int)((tiles + 7) / 8), 256, 0, stream>>>(Ob, WpT, bp, x, y);
    }
    // 5) layernorm
    k_layernorm<<<B_ * S_, 256, 0, stream>>>(y, lng, lnb, out);
}
